// HalfKPNetwork_57586921505234
// MI455X (gfx1250) — matrix operand order repaired and checked by reading, not compiled, not run on hardware
//
#include <hip/hip_runtime.h>

// ---- problem constants (from reference) ----
#define B_TOTAL  16384
#define H1       256       // per-side feature width
#define BAG      30
#define NPOS     64        // positions per block

// ---- LDS strides (halves / floats), padded to break bank conflicts ----
#define XS   520           // xh row stride in halves (512 + 8)
#define W1S  520           // w1h row stride in halves
#define W2S  40            // w2h row stride in halves (32 + 8)
#define Y1S  40            // y1 row stride in halves
#define Y2S  33            // y2 row stride in floats (32 + 1)

typedef _Float16 v4h  __attribute__((ext_vector_type(4)));
typedef _Float16 v8h  __attribute__((ext_vector_type(8)));
typedef _Float16 v16h __attribute__((ext_vector_type(16)));
typedef float    v8f  __attribute__((ext_vector_type(8)));

__device__ __forceinline__ v16h cat8(v8h lo, v8h hi) {
  return __builtin_shufflevector(lo, hi, 0,1,2,3,4,5,6,7,8,9,10,11,12,13,14,15);
}

__device__ __forceinline__ v4h cvt4(float a, float b, float c, float d) {
  v4h r;
  r.x = (_Float16)a; r.y = (_Float16)b; r.z = (_Float16)c; r.w = (_Float16)d;
  return r;
}

__global__ __launch_bounds__(256) void nnue_fused_kernel(
    const int*   __restrict__ widx,   // [B*BAG] int32
    const int*   __restrict__ bidx,   // [B*BAG] int32
    const float* __restrict__ ftw,    // [NUM_EMB, 256]
    const float* __restrict__ ftb,    // [NUM_EMB, 256]
    const float* __restrict__ fc1_w,  // [32, 512]
    const float* __restrict__ fc1_b,  // [32]
    const float* __restrict__ fc2_w,  // [32, 32]
    const float* __restrict__ fc2_b,  // [32]
    const float* __restrict__ fc3_w,  // [32]
    const float* __restrict__ fc3_b,  // [1]
    float*       __restrict__ out)    // [B]
{
  __shared__ _Float16 xh [NPOS * XS];   // activations (f16), [64][512] padded
  __shared__ _Float16 w1h[32 * W1S];    // fc1 weights as f16, [32][512] padded
  __shared__ _Float16 w2h[32 * W2S];    // fc2 weights as f16, [32][32] padded
  __shared__ _Float16 y1 [NPOS * Y1S];  // fc1 output (relu, f16), [64][32] padded
  __shared__ float    y2 [NPOS * Y2S];  // fc2 output (relu, f32), [64][32] padded

  const int tid  = threadIdx.x;
  const int pos0 = blockIdx.x * NPOS;

  // ---------- Phase 0: stage fc1_w / fc2_w into LDS as f16 ----------
  // fc1_w: 32*512 floats = 4096 float4 groups, 16 per thread
  #pragma unroll
  for (int g = 0; g < 16; ++g) {
    const int f = (tid * 16 + g) * 4;          // float index
    const int r = f >> 9, c = f & 511;
    const float4 v = *(const float4*)(fc1_w + f);
    *(v4h*)&w1h[r * W1S + c] = cvt4(v.x, v.y, v.z, v.w);
  }
  // fc2_w: 32*32 floats = 256 float4 groups, 1 per thread
  {
    const int f = tid * 4;
    const int r = f >> 5, c = f & 31;
    const float4 v = *(const float4*)(fc2_w + f);
    *(v4h*)&w2h[r * W2S + c] = cvt4(v.x, v.y, v.z, v.w);
  }

  // ---------- Phase 1: embedding-bag sum + ReLU -> xh (f16) ----------
  // 64 column-threads x 4 positions concurrent. Within a wave the position is
  // uniform, so force the row index into SGPRs (readfirstlane): the table load
  // becomes saddr-form global_load_b128 with a loop-invariant vector offset.
  const int col = (tid & 63) * 4;              // float column within 256
  const int sub = tid >> 6;                    // 0..3 (wave-uniform)
  for (int side = 0; side < 2; ++side) {
    const int*   __restrict__ idxp = side ? bidx : widx;
    const float* __restrict__ tab  = side ? ftb  : ftw;
    for (int p = sub; p < NPOS; p += 4) {
      const int base = (pos0 + p) * BAG;
      // hoist the 30 bag indices into SGPRs
      int rows[BAG];
      #pragma unroll
      for (int j = 0; j < BAG; ++j)
        rows[j] = __builtin_amdgcn_readfirstlane(idxp[base + j]);
      float ax = 0.f, ay = 0.f, az = 0.f, aw = 0.f;
      #pragma unroll
      for (int j = 0; j < BAG; ++j) {
        const float4 v = *(const float4*)(tab + (size_t)rows[j] * H1 + col);
        ax += v.x; ay += v.y; az += v.z; aw += v.w;
      }
      *(v4h*)&xh[p * XS + side * H1 + col] =
          cvt4(fmaxf(ax, 0.f), fmaxf(ay, 0.f), fmaxf(az, 0.f), fmaxf(aw, 0.f));
    }
  }
  __syncthreads();

  // ---------- wave / tile mapping for WMMA ----------
  const int lane = tid & 31;
  const int wave = tid >> 5;         // 0..7
  const int mt   = wave >> 1;        // M tile 0..3  (16 rows each)
  const int nt   = wave & 1;         // N tile 0..1  (16 cols each)
  const int lh   = lane >> 4;        // half-wave select
  const int l15  = lane & 15;
  const int n    = nt * 16 + l15;    // output neuron for B/C/D
  const int am   = mt * 16 + l15;    // activation row for A

  // ---------- Phase 2: fc1 = relu(x @ fc1_w^T + b1) via WMMA ----------
  v8f c1;
  {
    const float bv = fc1_b[n];
    #pragma unroll
    for (int r = 0; r < 8; ++r) c1[r] = bv;
  }
  #pragma unroll 4
  for (int k0 = 0; k0 < 512; k0 += 32) {
    // A: 16x32 f16 — lanes 0-15: K[k0..k0+7]+K[k0+16..23]; lanes 16-31: +8
    const v8h alo = *(const v8h*)&xh[am * XS + k0 + lh * 8];
    const v8h ahi = *(const v8h*)&xh[am * XS + k0 + 16 + lh * 8];
    const v16h A = cat8(alo, ahi);
    // B: 32x16 f16 — column n, lanes 0-15: K[k0..k0+7]+K[k0+16..23]; lanes 16-31: +8
    const v8h blo = *(const v8h*)&w1h[n * W1S + k0 + lh * 8];
    const v8h bhi = *(const v8h*)&w1h[n * W1S + k0 + 16 + lh * 8];
    const v16h Bm = cat8(blo, bhi);
    c1 = __builtin_amdgcn_wmma_f32_16x16x32_f16(false, A, false, Bm,
                                                (short)0, c1, false, false);
  }
  // D layout: VGPR r -> row (mt*16 + r + 8*lh), col n
  #pragma unroll
  for (int r = 0; r < 8; ++r)
    y1[(mt * 16 + r + 8 * lh) * Y1S + n] = (_Float16)fmaxf(c1[r], 0.f);
  __syncthreads();

  // ---------- Phase 3: fc2 = relu(y1 @ fc2_w^T + b2) via WMMA ----------
  v8f c2;
  {
    const float bv = fc2_b[n];
    #pragma unroll
    for (int r = 0; r < 8; ++r) c2[r] = bv;
  }
  {
    const v8h alo = *(const v8h*)&y1[am * Y1S + lh * 8];
    const v8h ahi = *(const v8h*)&y1[am * Y1S + 16 + lh * 8];
    const v16h A  = cat8(alo, ahi);
    const v8h blo = *(const v8h*)&w2h[n * W2S + lh * 8];
    const v8h bhi = *(const v8h*)&w2h[n * W2S + 16 + lh * 8];
    const v16h Bm = cat8(blo, bhi);
    c2 = __builtin_amdgcn_wmma_f32_16x16x32_f16(false, A, false, Bm,
                                                (short)0, c2, false, false);
  }
  #pragma unroll
  for (int r = 0; r < 8; ++r)
    y2[(mt * 16 + r + 8 * lh) * Y2S + n] = fmaxf(c2[r], 0.f);
  __syncthreads();

  // ---------- Phase 4: fc3 (dot over 32) ----------
  if (tid < NPOS) {
    float w3[32];
    #pragma unroll
    for (int j = 0; j < 32; ++j) w3[j] = fc3_w[j];
    float acc = fc3_b[0];
    #pragma unroll
    for (int j = 0; j < 32; ++j)
      acc += y2[tid * Y2S + j] * w3[j];
    out[pos0 + tid] = acc;
  }
}

extern "C" void kernel_launch(void* const* d_in, const int* in_sizes, int n_in,
                              void* d_out, int out_size, void* d_ws, size_t ws_size,
                              hipStream_t stream) {
  (void)in_sizes; (void)n_in; (void)out_size; (void)d_ws; (void)ws_size;
  const int*   widx = (const int*)  d_in[0];
  // d_in[1] = white_offsets (arange*BAG) — uniform bags, not needed
  const int*   bidx = (const int*)  d_in[2];
  // d_in[3] = black_offsets — not needed
  const float* ftw  = (const float*)d_in[4];
  const float* ftb  = (const float*)d_in[5];
  const float* fc1w = (const float*)d_in[6];
  const float* fc1b = (const float*)d_in[7];
  const float* fc2w = (const float*)d_in[8];
  const float* fc2b = (const float*)d_in[9];
  const float* fc3w = (const float*)d_in[10];
  const float* fc3b = (const float*)d_in[11];
  float* out = (float*)d_out;

  nnue_fused_kernel<<<B_TOTAL / NPOS, 256, 0, stream>>>(
      widx, bidx, ftw, ftb, fc1w, fc1b, fc2w, fc2b, fc3w, fc3b, out);
}
